// _ScratchXLSTM_6640019439855
// MI455X (gfx1250) — matrix operand order repaired and checked by reading, not compiled, not run on hardware
//
#include <hip/hip_runtime.h>
#include <hip/hip_bf16.h>

// B=2048 S=64 D=128 V=256 H=4 d=32 NB=2, T=B*S=131072 tokens.
// All GEMMs via v_wmma_f32_16x16x32_f16; all operands packed for b128 loads.
//
// B-operand packed layout: [tile][lane 0..31][16 halves] where entry
//   [tile][L][i] = B[kb(L)+i+(i&8)][n(L)], kb = (L>=16)*8, n = L&15   (ISA 7.12.2)
// A-operand packed layout ("apk"): [ktile][lane 0..31][16 halves] where for a
//   row-major 16x128 tile, element (row r, col c) lives at
//   [c>>5][r + ((c>>3)&1)*16][ ((c&16)>>1) + ((c&7)>>1)*2 + (c&1) ]
//   -> 8 consecutive cols (8-aligned) = 8 consecutive slots (vector stores ok)
// C-tile layout ("wxp"): [col][16 rows] -> lane loads rows mo..mo+7 contiguous.

typedef __attribute__((ext_vector_type(16))) _Float16 v16h;
typedef __attribute__((ext_vector_type(8)))  float    v8f;

#define WMMA(a, b, c) \
  __builtin_amdgcn_wmma_f32_16x16x32_f16(false, (a), false, (b), (short)0, (c), false, false)

struct __align__(8)  half4s { _Float16 v[4]; };
struct __align__(16) half8s { _Float16 v[8]; };

__device__ inline float wave_sum(float v) {
#pragma unroll
  for (int m = 16; m > 0; m >>= 1) v += __shfl_xor(v, m, 32);
  return v;
}

// ---- 1. pack f32 weight matrices into f16 B-operand tiles ----------------
// src: nmat row-major KxN matrices, mat_stride apart. One wave per 32x16 tile.
// dst tile order: mat-major, then ktile-major, then ntile.
__global__ void pack_b_kernel(const float* __restrict__ src, _Float16* __restrict__ dst,
                              int K, int N, int mat_stride) {
  const int lane = threadIdx.x & 31;
  const int tile = blockIdx.x * 8 + (threadIdx.x >> 5);
  const int tilesN = N >> 4;
  const int tpm = (K >> 5) * tilesN;
  const int mat = tile / tpm, rem = tile % tpm;
  const int kt = rem / tilesN, nt = rem % tilesN;
  const float* s = src + (size_t)mat * mat_stride +
                   (size_t)(kt * 32 + ((lane >> 4) << 3)) * N + nt * 16 + (lane & 15);
  _Float16* d = dst + ((size_t)tile * 32 + lane) * 16;
#pragma unroll
  for (int i = 0; i < 16; ++i) d[i] = (_Float16)s[(size_t)(i + (i & 8)) * N];
}

// ---- 2. embedding gather: one wave per token, float4 per lane ------------
__global__ void embed_kernel(const int* __restrict__ x, const float* __restrict__ em,
                             float* __restrict__ h) {
  const int lane = threadIdx.x & 31, w = threadIdx.x >> 5;
  const size_t t = (size_t)blockIdx.x * 8 + w;
  const int idx = x[t];
  ((float4*)(h + t * 128))[lane] = ((const float4*)(em + (size_t)idx * 128))[lane];
}

// ---- 3a. LayerNorm -> row-major f16 (consumed by batch-tiled wx_kernel) --
__global__ void ln_kernel(const float* __restrict__ h, const float* __restrict__ w,
                          _Float16* __restrict__ out) {
  const int lane = threadIdx.x & 31, wv = threadIdx.x >> 5;
  const size_t t = (size_t)blockIdx.x * 8 + wv;
  float4 v = ((const float4*)(h + t * 128))[lane];
  float mu = wave_sum(v.x + v.y + v.z + v.w) * (1.f / 128.f);
  float dx = v.x - mu, dy = v.y - mu, dz = v.z - mu, dw = v.w - mu;
  float rstd = rsqrtf(wave_sum(dx * dx + dy * dy + dz * dz + dw * dw) * (1.f / 128.f) + 1e-5f);
  float4 g = ((const float4*)w)[lane];
  half4s o;
  o.v[0] = (_Float16)(dx * rstd * g.x);
  o.v[1] = (_Float16)(dy * rstd * g.y);
  o.v[2] = (_Float16)(dz * rstd * g.z);
  o.v[3] = (_Float16)(dw * rstd * g.w);
  ((half4s*)(out + t * 128))[lane] = o;
}

// ---- 3b. LayerNorm -> A-packed f16 (token-tiled; consumed by proj) -------
__global__ void ln_packed_kernel(const float* __restrict__ h, const float* __restrict__ w,
                                 _Float16* __restrict__ outp) {
  const int lane = threadIdx.x & 31, wv = threadIdx.x >> 5;
  const size_t t = (size_t)blockIdx.x * 8 + wv;
  float4 v = ((const float4*)(h + t * 128))[lane];
  float mu = wave_sum(v.x + v.y + v.z + v.w) * (1.f / 128.f);
  float dx = v.x - mu, dy = v.y - mu, dz = v.z - mu, dw = v.w - mu;
  float rstd = rsqrtf(wave_sum(dx * dx + dy * dy + dz * dz + dw * dw) * (1.f / 128.f) + 1e-5f);
  float4 g = ((const float4*)w)[lane];
  half4s o;
  o.v[0] = (_Float16)(dx * rstd * g.x);
  o.v[1] = (_Float16)(dy * rstd * g.y);
  o.v[2] = (_Float16)(dz * rstd * g.z);
  o.v[3] = (_Float16)(dw * rstd * g.w);
  const int c = lane << 2;                 // cols c..c+3
  const int tile = (int)(t >> 4), r = (int)(t & 15);
  const int kt = c >> 5, hi = (c >> 3) & 1;
  const int sl = ((c & 16) >> 1) + ((c & 7) >> 1) * 2;
  *(half4s*)&outp[(((size_t)tile * 4 + kt) * 32 + r + hi * 16) * 16 + sl] = o;
}

// ---- 4. sLSTM input projection, batch-tiled at fixed s -------------------
// grid = 64 s * 128 btiles; wave w: head hh=w>>1, e-half eh=w&1.
// Output wxp in C-tile layout [s][btile][col 0..511][16 rows] (16B stores).
__global__ void wx_kernel(const _Float16* __restrict__ xn,
                          const _Float16* __restrict__ wgp,   // per-l B-packed
                          const float* __restrict__ bias,     // (4,128) f32
                          _Float16* __restrict__ wxp) {
  __shared__ __align__(32) _Float16 apk[4][32][16];
  const int tid = threadIdx.x, lane = tid & 31, w = tid >> 5;
  const int hh = w >> 1, eh = w & 1;
  const int s = blockIdx.x >> 7, btile = blockIdx.x & 127;
  const int b0 = btile * 16;
  {  // stage 16 batches x 128 cols (strided rows) into A-packed LDS
    const int r = tid >> 4, c0 = (tid & 15) << 3;
    float4 v = *(const float4*)(xn + ((size_t)(b0 + r) * 64 + s) * 128 + c0);
    const int kt = c0 >> 5, hi = (c0 >> 3) & 1, sl = (c0 & 16) >> 1;
    *(float4*)&apk[kt][r + hi * 16][sl] = v;
  }
  __syncthreads();
  const int nn = lane & 15, mo = (lane >> 4) << 3;
  v16h a = ((const v16h*)&apk[hh][0][0])[lane];
#pragma unroll
  for (int g = 0; g < 4; ++g) {
    v16h bm = ((const v16h*)wgp)[(size_t)(((g * 4 + hh) * 2 + eh) * 32 + lane)];
    v8f c = {};
    c = WMMA(a, bm, c);
    const int col = g * 128 + hh * 32 + eh * 16 + nn;
    const float bv = bias[col];
    half8s o;
#pragma unroll
    for (int j = 0; j < 8; ++j) o.v[j] = (_Float16)(c[j] + bv);
    *(half8s*)(wxp + (((size_t)s * 128 + btile) * 512 + col) * 16 + mo) = o;
  }
}

// ---- 5. sequential sLSTM scan: 16 batches/block, 64 steps ----------------
__global__ void scan_kernel(const _Float16* __restrict__ wxp,
                            const _Float16* __restrict__ rp,  // per-l B-packed
                            float* __restrict__ yout) {       // (B*S,128) f32
  __shared__ __align__(32) _Float16 ylds[2][4][32][16];       // A-packed y
  const int tid = threadIdx.x, lane = tid & 31, w = tid >> 5;
  const int hh = w >> 1, eh = w & 1;
  const int btile = blockIdx.x, b0 = btile * 16;
  const int nn = lane & 15, mo = (lane >> 4) << 3;
  const int e = eh * 16 + nn;            // col within head (0..31)
  const int colh = hh * 32 + e;          // col within 128
  const int hiw = (e >> 3) & 1;
  const int slot = ((e & 16) >> 1) + ((e & 7) >> 1) * 2 + (e & 1);

  v16h rb[4];
#pragma unroll
  for (int g = 0; g < 4; ++g)
    rb[g] = ((const v16h*)rp)[(size_t)((hh * 8 + g * 2 + eh) * 32 + lane)];

  ((float4*)&ylds[0][0][0][0])[tid] = make_float4(0.f, 0.f, 0.f, 0.f);
  __syncthreads();

  v8f cs = {}, ns = {}, ms;
#pragma unroll
  for (int j = 0; j < 8; ++j) ms[j] = -1e30f;

  int buf = 0;
  for (int s = 0; s < 64; ++s) {
    const _Float16* wtile = wxp + (((size_t)s * 128 + btile) * 512 + colh) * 16 + mo;
    if (s + 1 < 64) {
#pragma unroll
      for (int g = 0; g < 4; ++g)   // next timestep's 4 gate lines
        __builtin_prefetch(wtile + (size_t)1048576 + g * 2048, 0, 1);
    }
    v8f raw[4];
#pragma unroll
    for (int g = 0; g < 4; ++g) {
      half8s hv = *(const half8s*)(wtile + (size_t)g * 2048);
#pragma unroll
      for (int j = 0; j < 8; ++j) raw[g][j] = (float)hv.v[j];
    }
    v16h a = ((const v16h*)&ylds[buf][hh][0][0])[lane];
#pragma unroll
    for (int g = 0; g < 4; ++g) raw[g] = WMMA(a, rb[g], raw[g]);

#pragma unroll
    for (int j = 0; j < 8; ++j) {
      float iraw = raw[0][j], fraw = raw[1][j], zraw = raw[2][j], oraw = raw[3][j];
      float lsig = fminf(fraw, 0.f) - log1pf(__expf(-fabsf(fraw)));
      float lfm = ms[j] + lsig;
      float mn = fmaxf(iraw, lfm);
      float ig = __expf(iraw - mn);
      float fg = __expf(lfm - mn);
      float cn = fg * cs[j] + ig * tanhf(zraw);
      float nn2 = fg * ns[j] + ig;
      float yv = (cn / nn2) / (1.f + __expf(-oraw));
      cs[j] = cn; ns[j] = nn2; ms[j] = mn;
      ylds[buf ^ 1][hh][mo + j + hiw * 16][slot] = (_Float16)yv;
      yout[((size_t)(b0 + mo + j) * 64 + s) * 128 + colh] = yv;
    }
    __syncthreads();
    buf ^= 1;
  }
}

// ---- 6. headnorm + residual into h, fused LN2 -> z (A-packed) ------------
__global__ void hn_kernel(const float* __restrict__ y, const float* __restrict__ gnw,
                          const float* __restrict__ ln2w, float* __restrict__ h,
                          _Float16* __restrict__ zp) {
  const int lane = threadIdx.x & 31, w = threadIdx.x >> 5;
  const size_t t = (size_t)blockIdx.x * 8 + w;
  float r[4];
#pragma unroll
  for (int hh = 0; hh < 4; ++hh) {
    const size_t o = t * 128 + hh * 32 + lane;
    float v = y[o];
    float mu = wave_sum(v) * (1.f / 32.f);
    float d0 = v - mu;
    float var = wave_sum(d0 * d0) * (1.f / 32.f);
    r[hh] = h[o] + d0 * rsqrtf(var + 1e-5f) * gnw[hh * 32 + lane];
    h[o] = r[hh];
  }
  float mu = wave_sum(r[0] + r[1] + r[2] + r[3]) * (1.f / 128.f);
  float q = 0.f;
#pragma unroll
  for (int hh = 0; hh < 4; ++hh) { float d = r[hh] - mu; q += d * d; }
  float rstd = rsqrtf(wave_sum(q) * (1.f / 128.f) + 1e-5f);
  const int tile = (int)(t >> 4), rr = (int)(t & 15);
  const int hi = (lane >> 3) & 1;
  const int sl = ((lane & 16) >> 1) + ((lane & 7) >> 1) * 2 + (lane & 1);
#pragma unroll
  for (int hh = 0; hh < 4; ++hh)
    zp[(((size_t)tile * 4 + hh) * 32 + rr + hi * 16) * 16 + sl] =
        (_Float16)((r[hh] - mu) * rstd * ln2w[hh * 32 + lane]);
}

// ---- 7. FFN up (128->256) + exact-GELU gating -> gg (A-packed) -----------
__global__ void up_kernel(const _Float16* __restrict__ zp,
                          const _Float16* __restrict__ wup,  // per-l B-packed
                          _Float16* __restrict__ ggp) {
  const int lane = threadIdx.x & 31, w = threadIdx.x >> 5;
  v8f cg = {}, cl = {};
#pragma unroll
  for (int kt = 0; kt < 4; ++kt) {
    v16h a = ((const v16h*)(zp + ((size_t)blockIdx.x * 4 + kt) * 512))[lane];
    v16h bg = ((const v16h*)wup)[(size_t)((kt * 16 + w) * 32 + lane)];
    v16h bl = ((const v16h*)wup)[(size_t)((kt * 16 + w + 8) * 32 + lane)];
    cg = WMMA(a, bg, cg);
    cl = WMMA(a, bl, cl);
  }
  const int nn = lane & 15, mo = (lane >> 4) << 3;
  const int e2 = ((w & 1) << 4) + nn, kt2 = w >> 1;
  const int hi2 = (e2 >> 3) & 1;
  const int sl2 = ((e2 & 16) >> 1) + ((e2 & 7) >> 1) * 2 + (e2 & 1);
#pragma unroll
  for (int j = 0; j < 8; ++j) {
    float xg = cg[j];
    float ge = 0.5f * xg * (1.f + erff(xg * 0.7071067811865476f));
    ggp[(((size_t)blockIdx.x * 4 + kt2) * 32 + mo + j + hi2 * 16) * 16 + sl2] =
        (_Float16)(ge * cl[j]);
  }
}

// ---- 8. FFN down (128->128) + residual into h; no LDS --------------------
__global__ void down_kernel(const _Float16* __restrict__ ggp,
                            const _Float16* __restrict__ wdp,  // per-l B-packed
                            float* __restrict__ h) {
  const int lane = threadIdx.x & 31, w = threadIdx.x >> 5;
  const int t0 = blockIdx.x * 16;
  const int nn = lane & 15, mo = (lane >> 4) << 3;
  v8f c;
#pragma unroll
  for (int j = 0; j < 8; ++j) c[j] = h[(size_t)(t0 + mo + j) * 128 + w * 16 + nn];
#pragma unroll
  for (int kt = 0; kt < 4; ++kt) {
    v16h a = ((const v16h*)(ggp + ((size_t)blockIdx.x * 4 + kt) * 512))[lane];
    v16h b = ((const v16h*)wdp)[(size_t)((kt * 8 + w) * 32 + lane)];
    c = WMMA(a, b, c);
  }
#pragma unroll
  for (int j = 0; j < 8; ++j) h[(size_t)(t0 + mo + j) * 128 + w * 16 + nn] = c[j];
}

// ---- 9. vocab projection (128->256) + bias, f32 out; no LDS --------------
__global__ void proj_kernel(const _Float16* __restrict__ xnp,
                            const _Float16* __restrict__ pwp,  // B-packed
                            const float* __restrict__ pb, float* __restrict__ out) {
  const int lane = threadIdx.x & 31, w = threadIdx.x >> 5;
  const int t0 = blockIdx.x * 16;
  const int nn = lane & 15, mo = (lane >> 4) << 3;
  const float b0v = pb[w * 16 + nn], b1v = pb[w * 16 + 128 + nn];
  v8f c0, c1;
#pragma unroll
  for (int j = 0; j < 8; ++j) { c0[j] = b0v; c1[j] = b1v; }
#pragma unroll
  for (int kt = 0; kt < 4; ++kt) {
    v16h a = ((const v16h*)(xnp + ((size_t)blockIdx.x * 4 + kt) * 512))[lane];
    v16h ba = ((const v16h*)pwp)[(size_t)((kt * 16 + w) * 32 + lane)];
    v16h bb = ((const v16h*)pwp)[(size_t)((kt * 16 + w + 8) * 32 + lane)];
    c0 = WMMA(a, ba, c0);
    c1 = WMMA(a, bb, c1);
  }
#pragma unroll
  for (int j = 0; j < 8; ++j) {
    out[(size_t)(t0 + mo + j) * 256 + w * 16 + nn] = c0[j];
    out[(size_t)(t0 + mo + j) * 256 + w * 16 + 128 + nn] = c1[j];
  }
}

extern "C" void kernel_launch(void* const* d_in, const int* in_sizes, int n_in,
                              void* d_out, int out_size, void* d_ws, size_t ws_size,
                              hipStream_t stream) {
  (void)in_sizes; (void)n_in; (void)out_size; (void)ws_size;
  const int* x = (const int*)d_in[0];
  const float* embed = (const float*)d_in[1];
  const float* Wg = (const float*)d_in[2];
  const float* R = (const float*)d_in[3];
  const float* bias = (const float*)d_in[4];
  const float* ln1w = (const float*)d_in[5];
  const float* gnw = (const float*)d_in[6];
  const float* ln2w = (const float*)d_in[7];
  const float* Wup = (const float*)d_in[8];
  const float* Wdn = (const float*)d_in[9];
  const float* postw = (const float*)d_in[10];
  const float* projW = (const float*)d_in[11];
  const float* projb = (const float*)d_in[12];
  float* out = (float*)d_out;

  const size_t T = (size_t)2048 * 64;
  char* ws = (char*)d_ws;
  float* hbuf = (float*)ws;      ws += T * 128 * 4;       // 67 MB
  _Float16* xn = (_Float16*)ws;  ws += T * 128 * 2;       // 33.5 MB (rowmajor or packed)
  _Float16* wxp = (_Float16*)ws; ws += T * 512 * 2;       // 134 MB, C-tile layout
  float* ybuf = (float*)ws;                               // 67 MB, y f32
  _Float16* ggp = (_Float16*)ybuf;                        //  reused as gg A-packed f16
  ws += T * 128 * 4;
  _Float16* wgp = (_Float16*)ws;  ws += 32768 * 2;  // 64 tiles
  _Float16* rp = (_Float16*)ws;   ws += 32768 * 2;  // 64 tiles
  _Float16* wupp = (_Float16*)ws; ws += 65536 * 2;  // 128 tiles
  _Float16* wdnp = (_Float16*)ws; ws += 32768 * 2;  // 64 tiles
  _Float16* pwp = (_Float16*)ws;  ws += 32768 * 2;  // 64 tiles

  // Wg: 32 mats of 32x32 (l,g,h); R: 8 mats of 32x128 (l,h);
  // Wup: 2 mats 128x256; Wdn: 2 mats 128x128; proj: 1 mat 128x256.
  pack_b_kernel<<<8, 256, 0, stream>>>(Wg, wgp, 32, 32, 1024);
  pack_b_kernel<<<8, 256, 0, stream>>>(R, rp, 32, 128, 4096);
  pack_b_kernel<<<16, 256, 0, stream>>>(Wup, wupp, 128, 256, 32768);
  pack_b_kernel<<<8, 256, 0, stream>>>(Wdn, wdnp, 128, 128, 16384);
  pack_b_kernel<<<8, 256, 0, stream>>>(projW, pwp, 128, 256, 32768);

  embed_kernel<<<T / 8, 256, 0, stream>>>(x, embed, hbuf);

  for (int l = 0; l < 2; ++l) {
    ln_kernel<<<T / 8, 256, 0, stream>>>(hbuf, ln1w + l * 128, xn);
    wx_kernel<<<64 * 128, 256, 0, stream>>>(xn, wgp + l * 16384, bias + l * 512, wxp);
    scan_kernel<<<128, 256, 0, stream>>>(wxp, rp + l * 16384, ybuf);
    hn_kernel<<<T / 8, 256, 0, stream>>>(ybuf, gnw + l * 128, ln2w + l * 128, hbuf, xn);
    up_kernel<<<T / 16, 256, 0, stream>>>(xn, wupp + l * 32768, ggp);
    down_kernel<<<T / 16, 256, 0, stream>>>(ggp, wdnp + l * 16384, hbuf);
  }
  ln_packed_kernel<<<T / 8, 256, 0, stream>>>(hbuf, postw, xn);
  proj_kernel<<<T / 16, 256, 0, stream>>>(xn, pwp, projb, out);
}
